// GlobalBlock_14276471292724
// MI455X (gfx1250) — matrix operand order repaired and checked by reading, not compiled, not run on hardware
//
#include <hip/hip_runtime.h>

// ---------------------------------------------------------------------------
// Types for CDNA5 WMMA (wave32, 16x16x32 f16 -> f32)
// ---------------------------------------------------------------------------
typedef _Float16 h16;
typedef __attribute__((ext_vector_type(16))) _Float16 v16h;
typedef __attribute__((ext_vector_type(8)))  float    v8f;

#define C_DIM   384
#define NHEAD   12
#define B_SZ    16
#define HW      56
#define L_TOK   3136
#define M_TOK   50176
#define SCALE_F 0.1767766952966369f   // 32^-0.5

__device__ inline v8f vzero8() {
    v8f z = {0.f, 0.f, 0.f, 0.f, 0.f, 0.f, 0.f, 0.f};
    return z;
}

__device__ inline v8f wmma_f16(v16h a, v16h b, v8f c) {
    // D = A x B + C ; f16 inputs, f32 accumulate
    return __builtin_amdgcn_wmma_f32_16x16x32_f16(false, a, false, b,
                                                  (short)0, c, false, false);
}

__device__ inline unsigned pack2h(float a, float b) {
    union { h16 h[2]; unsigned u; } p;
    p.h[0] = (h16)a; p.h[1] = (h16)b;
    return p.u;
}

// A fragment (16x32, f16) per ISA layout: lane m=l&15, K-halves split by l>>4.
__device__ inline v16h frag_A_h16(const h16* base, int stride, int lane) {
    const int m  = lane & 15;
    const int kb = (lane >> 4) << 3;        // 0 or 8
    const h16* row = base + m * stride;
    union { v16h v; unsigned u[8]; } f;
#pragma unroll
    for (int i = 0; i < 4; ++i) {
        f.u[i]     = *(const unsigned*)(row + kb + 2 * i);
        f.u[i + 4] = *(const unsigned*)(row + 16 + kb + 2 * i);
    }
    return f.v;
}

// A fragment converted on the fly from f32 LDS rows (attention probabilities).
__device__ inline v16h frag_A_f32(const float* base, int stride, int lane) {
    const int m  = lane & 15;
    const int kb = (lane >> 4) << 3;
    const float* row = base + m * stride;
    union { v16h v; unsigned u[8]; } f;
#pragma unroll
    for (int i = 0; i < 4; ++i) {
        f.u[i]     = pack2h(row[kb + 2 * i],      row[kb + 2 * i + 1]);
        f.u[i + 4] = pack2h(row[16 + kb + 2 * i], row[16 + kb + 2 * i + 1]);
    }
    return f.v;
}

// B fragment (32x16, f16) from B^T storage (rows = n, cols = k):
// lane n=l&15, K-halves split by l>>4 (same order as A).
__device__ inline v16h frag_B_h16(const h16* base, int stride, int lane) {
    const int n  = lane & 15;
    const int kb = (lane >> 4) << 3;        // 0 or 8
    const h16* row = base + n * stride;
    union { v16h v; unsigned u[8]; } f;
#pragma unroll
    for (int i = 0; i < 4; ++i) {
        f.u[i]     = *(const unsigned*)(row + kb + 2 * i);
        f.u[i + 4] = *(const unsigned*)(row + 16 + kb + 2 * i);
    }
    return f.v;
}

// ---------------------------------------------------------------------------
// Generic 128x128 WMMA GEMM: C = A[MxK](f16) * B[KxN] (given as B^T f16 [N][K])
// 256 threads = 8 waves, each wave owns a 64x32 tile (4x2 WMMA tiles).
// Two-stage software pipeline: global->regs prefetch overlaps the WMMA burst,
// regs->LDS commit happens after compute; double-buffered LDS.
// ---------------------------------------------------------------------------
enum { EPI_BIAS_F16 = 0, EPI_AFFINE_F16 = 1, EPI_BIAS_RES_F32 = 2, EPI_BIAS_GELU_F16 = 3 };

#define GLDA 40   // padded LDS row stride (halves): conflict-free & 16B aligned

template<int EPI>
__global__ __launch_bounds__(256)
void gemm_wmma(const h16* __restrict__ A, const h16* __restrict__ BT,
               const float* __restrict__ bias, const float* __restrict__ gain,
               const float* __restrict__ resid, void* __restrict__ out,
               int M, int N, int K)
{
    __shared__ h16 As[2][128 * GLDA];
    __shared__ h16 Bs[2][128 * GLDA];

    const int tid  = threadIdx.x;
    const int lane = tid & 31;
    const int wid  = tid >> 5;
    const int wm   = wid >> 2;       // 0..1
    const int wn   = wid & 3;        // 0..3

    const int m0 = blockIdx.y * 128;
    const int n0 = blockIdx.x * 128;

    // per-thread staging coordinates (two 16B chunks of A and of B^T per step)
    int rA[2], rB[2], rr[2], qq[2];
#pragma unroll
    for (int i = 0; i < 2; ++i) {
        int idx = tid + i * 256;         // 0..511
        rr[i] = idx >> 2;                // tile row 0..127
        qq[i] = idx & 3;                 // 8-half chunk
        int gr = m0 + rr[i]; if (gr >= M) gr = M - 1;
        rA[i] = gr;
        int gn = n0 + rr[i]; if (gn >= N) gn = N - 1;
        rB[i] = gn;
    }

    // prologue: stage K-step 0 into buffer 0
#pragma unroll
    for (int i = 0; i < 2; ++i) {
        *(uint4*)(As[0] + rr[i] * GLDA + qq[i] * 8) =
            *(const uint4*)(A + (size_t)rA[i] * K + qq[i] * 8);
        *(uint4*)(Bs[0] + rr[i] * GLDA + qq[i] * 8) =
            *(const uint4*)(BT + (size_t)rB[i] * K + qq[i] * 8);
    }
    __syncthreads();

    v8f acc[4][2];
#pragma unroll
    for (int i = 0; i < 4; ++i) { acc[i][0] = vzero8(); acc[i][1] = vzero8(); }

    const int NK = K >> 5;
    for (int ks = 0; ks < NK; ++ks) {
        const int cur  = ks & 1;
        const bool more = (ks + 1) < NK;

        // issue next tile's global loads (latency hidden under WMMA below)
        uint4 pa[2], pb[2];
        if (more) {
            const int k0 = (ks + 1) << 5;
#pragma unroll
            for (int i = 0; i < 2; ++i) {
                pa[i] = *(const uint4*)(A  + (size_t)rA[i] * K + k0 + qq[i] * 8);
                pb[i] = *(const uint4*)(BT + (size_t)rB[i] * K + k0 + qq[i] * 8);
            }
        }

        // matrix math on current buffer
        v16h bf0 = frag_B_h16(Bs[cur] + (wn * 32 + 0)  * GLDA, GLDA, lane);
        v16h bf1 = frag_B_h16(Bs[cur] + (wn * 32 + 16) * GLDA, GLDA, lane);
#pragma unroll
        for (int tm = 0; tm < 4; ++tm) {
            v16h af = frag_A_h16(As[cur] + (wm * 64 + tm * 16) * GLDA, GLDA, lane);
            acc[tm][0] = wmma_f16(af, bf0, acc[tm][0]);
            acc[tm][1] = wmma_f16(af, bf1, acc[tm][1]);
        }

        // commit prefetched tile into the other buffer
        if (more) {
            const int nxt = cur ^ 1;
#pragma unroll
            for (int i = 0; i < 2; ++i) {
                *(uint4*)(As[nxt] + rr[i] * GLDA + qq[i] * 8) = pa[i];
                *(uint4*)(Bs[nxt] + rr[i] * GLDA + qq[i] * 8) = pb[i];
            }
        }
        __syncthreads();
    }

    const int ln_ = lane & 15;
    const int lh  = lane >> 4;
#pragma unroll
    for (int tm = 0; tm < 4; ++tm)
#pragma unroll
    for (int tn = 0; tn < 2; ++tn)
#pragma unroll
    for (int v = 0; v < 8; ++v) {
        int row = m0 + wm * 64 + tm * 16 + v + 8 * lh;
        int col = n0 + wn * 32 + tn * 16 + ln_;
        if (row < M) {
            float a = acc[tm][tn][v];
            size_t o = (size_t)row * N + col;
            if (EPI == EPI_BIAS_F16) {
                ((h16*)out)[o] = (h16)(a + bias[col]);
            } else if (EPI == EPI_AFFINE_F16) {
                ((h16*)out)[o] = (h16)(a * gain[col] + bias[col]);
            } else if (EPI == EPI_BIAS_RES_F32) {
                ((float*)out)[o] = a + bias[col] + resid[o];
            } else { // exact GELU
                float xg = a + bias[col];
                ((h16*)out)[o] = (h16)(0.5f * xg * (1.0f + erff(xg * 0.7071067811865475f)));
            }
        }
    }
}

// ---------------------------------------------------------------------------
// Windowed attention: one wave per (head, window, batch).
// 64 queries, 128 keys (64 window + 64 global), RPB on first 64 keys.
// ---------------------------------------------------------------------------
__global__ __launch_bounds__(32)
void win_attn(const h16* __restrict__ qkv,   // [M_TOK, 1152] f16
              const h16* __restrict__ gk,    // [B*64, 384] f16
              const h16* __restrict__ gv,    // [B*64, 384] f16
              const float* __restrict__ rpb, // [225, 12]
              h16* __restrict__ ow)          // [M_TOK, 384] f16
{
    const int h  = blockIdx.x;
    const int w  = blockIdx.y;
    const int b  = blockIdx.z;
    const int wr = w / 7, wc = w - wr * 7;
    const int lane = threadIdx.x;

    __shared__ h16   Ks[128 * 36];   // keys   [key][hd]   (= B^T for Q*K^T)
    __shared__ h16   VTs[32 * 132];  // values [hd][key]   (= B^T for P*V)
    __shared__ float S[64 * 129];    // logits / probabilities

    const size_t tokbase = (size_t)b * L_TOK + (wr * 8) * HW + wc * 8;

    for (int key = lane; key < 128; key += 32) {
        const h16* ksrc; const h16* vsrc;
        if (key < 64) {
            size_t l = tokbase + (size_t)(key >> 3) * HW + (key & 7);
            ksrc = qkv + l * 1152 + 384 + h * 32;
            vsrc = qkv + l * 1152 + 768 + h * 32;
        } else {
            size_t r = (size_t)b * 64 + (key - 64);
            ksrc = gk + r * C_DIM + h * 32;
            vsrc = gv + r * C_DIM + h * 32;
        }
        h16* kd = Ks + key * 36;
#pragma unroll
        for (int i = 0; i < 4; ++i) ((uint2*)kd)[i] = ((const uint2*)ksrc)[i];
#pragma unroll
        for (int c = 0; c < 32; ++c) VTs[c * 132 + key] = vsrc[c];
    }
    __syncthreads();

    // Q fragments straight from global (per-lane row addressing)
    v16h qf[4];
    {
        const int m  = lane & 15;
        const int kb = (lane >> 4) << 3;
#pragma unroll
        for (int tm = 0; tm < 4; ++tm) {
            int pos = tm * 16 + m;
            size_t l = tokbase + (size_t)(pos >> 3) * HW + (pos & 7);
            const h16* row = qkv + l * 1152 + h * 32;
            union { v16h v; unsigned u[8]; } f;
#pragma unroll
            for (int i = 0; i < 4; ++i) {
                f.u[i]     = *(const unsigned*)(row + kb + 2 * i);
                f.u[i + 4] = *(const unsigned*)(row + 16 + kb + 2 * i);
            }
            qf[tm] = f.v;
        }
    }

    const int ln_ = lane & 15;
    const int lh  = lane >> 4;

    // S = Q * K^T (64x128, K=32) -> 32 WMMAs
#pragma unroll
    for (int tn = 0; tn < 8; ++tn) {
        v16h bf = frag_B_h16(Ks + tn * 16 * 36, 36, lane);
#pragma unroll
        for (int tm = 0; tm < 4; ++tm) {
            v8f c = wmma_f16(qf[tm], bf, vzero8());
#pragma unroll
            for (int v = 0; v < 8; ++v)
                S[(tm * 16 + v + 8 * lh) * 129 + tn * 16 + ln_] = c[v];
        }
    }
    __syncthreads();

    // softmax rows (lane handles rows lane and lane+32); scale + RPB on local keys
    for (int rr = 0; rr < 2; ++rr) {
        const int r = lane + rr * 32;
        const int qi = r >> 3, qj = r & 7;
        float* Sr = S + r * 129;
        float mx = -1e30f;
        for (int c = 0; c < 128; ++c) {
            float s = Sr[c] * SCALE_F;
            if (c < 64) {
                int ki = c >> 3, kj = c & 7;
                s += rpb[((qi - ki + 7) * 15 + (qj - kj + 7)) * NHEAD + h];
            }
            Sr[c] = s;
            mx = fmaxf(mx, s);
        }
        float sum = 0.f;
        for (int c = 0; c < 128; ++c) { float e = __expf(Sr[c] - mx); Sr[c] = e; sum += e; }
        float inv = 1.f / sum;
        for (int c = 0; c < 128; ++c) Sr[c] *= inv;
    }
    __syncthreads();

    // O = P * V (64x32, K=128) -> 32 WMMAs, P fragments packed from f32 LDS
    v8f acc[4][2];
#pragma unroll
    for (int i = 0; i < 4; ++i) { acc[i][0] = vzero8(); acc[i][1] = vzero8(); }
#pragma unroll
    for (int ks = 0; ks < 4; ++ks) {
        v16h bf0 = frag_B_h16(VTs + 0  * 132 + ks * 32, 132, lane);
        v16h bf1 = frag_B_h16(VTs + 16 * 132 + ks * 32, 132, lane);
#pragma unroll
        for (int tm = 0; tm < 4; ++tm) {
            v16h af = frag_A_f32(S + (tm * 16) * 129 + ks * 32, 129, lane);
            acc[tm][0] = wmma_f16(af, bf0, acc[tm][0]);
            acc[tm][1] = wmma_f16(af, bf1, acc[tm][1]);
        }
    }
#pragma unroll
    for (int tm = 0; tm < 4; ++tm)
#pragma unroll
    for (int tn = 0; tn < 2; ++tn)
#pragma unroll
    for (int v = 0; v < 8; ++v) {
        int pos = tm * 16 + v + 8 * lh;
        size_t l = tokbase + (size_t)(pos >> 3) * HW + (pos & 7);
        ow[l * C_DIM + h * 32 + tn * 16 + ln_] = (h16)acc[tm][tn][v];
    }
}

// ---------------------------------------------------------------------------
// Cross attention (learnable queries over 49 pooled tokens, zero-padded to 64)
// Writes oq = attn_out + q (f16), one wave per (head, batch).
// ---------------------------------------------------------------------------
__global__ __launch_bounds__(32)
void cross_attn(const h16* __restrict__ qT,    // [64, 384] f16
                const h16* __restrict__ keyb,  // [784, 384] f16
                const h16* __restrict__ valb,  // [784, 384] f16
                h16* __restrict__ oq)          // [1024, 384] f16
{
    const int h = blockIdx.x;
    const int b = blockIdx.y;
    const int lane = threadIdx.x;

    __shared__ h16   Ks[64 * 36];
    __shared__ h16   VTs[32 * 68];
    __shared__ float S[64 * 65];

    for (int key = lane; key < 64; key += 32) {
        h16* kd = Ks + key * 36;
        if (key < 49) {
            const h16* ksrc = keyb + ((size_t)b * 49 + key) * C_DIM + h * 32;
            const h16* vsrc = valb + ((size_t)b * 49 + key) * C_DIM + h * 32;
#pragma unroll
            for (int i = 0; i < 4; ++i) ((uint2*)kd)[i] = ((const uint2*)ksrc)[i];
#pragma unroll
            for (int c = 0; c < 32; ++c) VTs[c * 68 + key] = vsrc[c];
        } else {
#pragma unroll
            for (int c = 0; c < 32; ++c) { kd[c] = (h16)0.f; VTs[c * 68 + key] = (h16)0.f; }
        }
    }
    __syncthreads();

    v16h qf[4];
    {
        const int m  = lane & 15;
        const int kb = (lane >> 4) << 3;
#pragma unroll
        for (int tm = 0; tm < 4; ++tm) {
            const h16* row = qT + (size_t)(tm * 16 + m) * C_DIM + h * 32;
            union { v16h v; unsigned u[8]; } f;
#pragma unroll
            for (int i = 0; i < 4; ++i) {
                f.u[i]     = *(const unsigned*)(row + kb + 2 * i);
                f.u[i + 4] = *(const unsigned*)(row + 16 + kb + 2 * i);
            }
            qf[tm] = f.v;
        }
    }

    const int ln_ = lane & 15;
    const int lh  = lane >> 4;
#pragma unroll
    for (int tn = 0; tn < 4; ++tn) {
        v16h bf = frag_B_h16(Ks + tn * 16 * 36, 36, lane);
#pragma unroll
        for (int tm = 0; tm < 4; ++tm) {
            v8f c = wmma_f16(qf[tm], bf, vzero8());
#pragma unroll
            for (int v = 0; v < 8; ++v)
                S[(tm * 16 + v + 8 * lh) * 65 + tn * 16 + ln_] = c[v];
        }
    }
    __syncthreads();

    for (int rr = 0; rr < 2; ++rr) {
        const int r = lane + rr * 32;
        float* Sr = S + r * 65;
        float mx = -1e30f;
        for (int c = 0; c < 64; ++c) {
            float s = (c < 49) ? Sr[c] * SCALE_F : -1e30f;
            Sr[c] = s;
            mx = fmaxf(mx, s);
        }
        float sum = 0.f;
        for (int c = 0; c < 64; ++c) { float e = __expf(Sr[c] - mx); Sr[c] = e; sum += e; }
        float inv = 1.f / sum;
        for (int c = 0; c < 64; ++c) Sr[c] *= inv;
    }
    __syncthreads();

    v8f acc[4][2];
#pragma unroll
    for (int i = 0; i < 4; ++i) { acc[i][0] = vzero8(); acc[i][1] = vzero8(); }
#pragma unroll
    for (int ks = 0; ks < 2; ++ks) {
        v16h bf0 = frag_B_h16(VTs + 0  * 68 + ks * 32, 68, lane);
        v16h bf1 = frag_B_h16(VTs + 16 * 68 + ks * 32, 68, lane);
#pragma unroll
        for (int tm = 0; tm < 4; ++tm) {
            v16h af = frag_A_f32(S + (tm * 16) * 65 + ks * 32, 65, lane);
            acc[tm][0] = wmma_f16(af, bf0, acc[tm][0]);
            acc[tm][1] = wmma_f16(af, bf1, acc[tm][1]);
        }
    }
#pragma unroll
    for (int tm = 0; tm < 4; ++tm)
#pragma unroll
    for (int tn = 0; tn < 2; ++tn)
#pragma unroll
    for (int v = 0; v < 8; ++v) {
        int pos = tm * 16 + v + 8 * lh;
        int col = h * 32 + tn * 16 + ln_;
        float qv = (float)qT[(size_t)pos * C_DIM + col];
        oq[((size_t)b * 64 + pos) * C_DIM + col] = (h16)(acc[tm][tn][v] + qv);
    }
}

// ---------------------------------------------------------------------------
// Utility kernels
// ---------------------------------------------------------------------------
__global__ void transpose_w(const float* __restrict__ W, h16* __restrict__ WT,
                            int K, int N)  // W:[K][N] -> WT:[N][K] f16
{
    int idx = blockIdx.x * blockDim.x + threadIdx.x;
    if (idx >= K * N) return;
    int n = idx / K;
    int k = idx - n * K;
    WT[idx] = (h16)W[(size_t)k * N + n];
}

__global__ void cvt_h16(const float* __restrict__ in, h16* __restrict__ out, int n)
{
    int i = blockIdx.x * blockDim.x + threadIdx.x;
    if (i < n) out[i] = (h16)in[i];
}

__global__ void fold_bn(const float* __restrict__ pb, const float* __restrict__ g,
                        const float* __restrict__ bb, float* __restrict__ gain,
                        float* __restrict__ bias2)
{
    int i = blockIdx.x * blockDim.x + threadIdx.x;
    if (i < C_DIM) {
        float s = rsqrtf(1.0f + 1e-5f);
        gain[i]  = s * g[i];
        bias2[i] = pb[i] * s * g[i] + bb[i];
    }
}

__global__ __launch_bounds__(32)
void ln_h16(const float* __restrict__ x, const float* __restrict__ g,
            const float* __restrict__ b, h16* __restrict__ out)
{
    const int row = blockIdx.x;
    const float* xr = x + (size_t)row * C_DIM;
    const int lane = threadIdx.x;
    float s = 0.f;
    for (int c = lane; c < C_DIM; c += 32) s += xr[c];
#pragma unroll
    for (int off = 16; off > 0; off >>= 1) s += __shfl_xor(s, off, 32);
    const float mu = s * (1.0f / C_DIM);
    float v = 0.f;
    for (int c = lane; c < C_DIM; c += 32) { float d = xr[c] - mu; v += d * d; }
#pragma unroll
    for (int off = 16; off > 0; off >>= 1) v += __shfl_xor(v, off, 32);
    const float rstd = rsqrtf(v * (1.0f / C_DIM) + 1e-5f);
    for (int c = lane; c < C_DIM; c += 32)
        out[(size_t)row * C_DIM + c] = (h16)((xr[c] - mu) * rstd * g[c] + b[c]);
}

__global__ __launch_bounds__(384)
void pool8(const h16* __restrict__ xn, h16* __restrict__ pooled)
{
    const int bp = blockIdx.x;            // b*49 + p
    const int b = bp / 49, p = bp - b * 49;
    const int ph = p / 7, pw = p - ph * 7;
    const int c = threadIdx.x;
    float s = 0.f;
    for (int i = 0; i < 8; ++i)
        for (int j = 0; j < 8; ++j) {
            size_t l = (size_t)b * L_TOK + (ph * 8 + i) * HW + (pw * 8 + j);
            s += (float)xn[l * C_DIM + c];
        }
    pooled[(size_t)bp * C_DIM + c] = (h16)(s * (1.f / 64.f));
}

// ---------------------------------------------------------------------------
// Host orchestration
// ---------------------------------------------------------------------------
extern "C" void kernel_launch(void* const* d_in, const int* in_sizes, int n_in,
                              void* d_out, int out_size, void* d_ws, size_t ws_size,
                              hipStream_t stream)
{
    const float* x       = (const float*)d_in[0];
    const float* query   = (const float*)d_in[1];
    const float* kproj_w = (const float*)d_in[2];
    const float* kproj_b = (const float*)d_in[3];
    const float* kbn_g   = (const float*)d_in[4];
    const float* kbn_b   = (const float*)d_in[5];
    const float* vproj_w = (const float*)d_in[6];
    const float* vproj_b = (const float*)d_in[7];
    const float* vbn_g   = (const float*)d_in[8];
    const float* vbn_b   = (const float*)d_in[9];
    const float* ck_w    = (const float*)d_in[10];
    const float* ck_b    = (const float*)d_in[11];
    const float* cv_w    = (const float*)d_in[12];
    const float* cv_b    = (const float*)d_in[13];
    const float* qkv_w   = (const float*)d_in[14];
    const float* qkv_b   = (const float*)d_in[15];
    const float* proj_w  = (const float*)d_in[16];
    const float* proj_b  = (const float*)d_in[17];
    const float* rpb     = (const float*)d_in[18];
    const float* n1_g    = (const float*)d_in[19];
    const float* n1_b    = (const float*)d_in[20];
    const float* n2_g    = (const float*)d_in[21];
    const float* n2_b    = (const float*)d_in[22];
    const float* fc1_w   = (const float*)d_in[23];
    const float* fc1_b   = (const float*)d_in[24];
    const float* fc2_w   = (const float*)d_in[25];
    const float* fc2_b   = (const float*)d_in[26];

    char* ws = (char*)d_ws;
    auto alloc = [&](size_t bytes) -> void* {
        char* p = ws;
        ws += (bytes + 255) & ~(size_t)255;
        return (void*)p;
    };

    h16* qkvT   = (h16*)alloc((size_t)1152 * 384 * 2);
    h16* projT  = (h16*)alloc((size_t)384 * 384 * 2);
    h16* fc1T   = (h16*)alloc((size_t)1536 * 384 * 2);
    h16* fc2T   = (h16*)alloc((size_t)384 * 1536 * 2);
    h16* kprojT = (h16*)alloc((size_t)384 * 384 * 2);
    h16* vprojT = (h16*)alloc((size_t)384 * 384 * 2);
    h16* ckT    = (h16*)alloc((size_t)384 * 384 * 2);
    h16* cvT    = (h16*)alloc((size_t)384 * 384 * 2);
    h16* queryT = (h16*)alloc((size_t)64 * 384 * 2);
    float* kgain = (float*)alloc(384 * 4);
    float* kbias = (float*)alloc(384 * 4);
    float* vgain = (float*)alloc(384 * 4);
    float* vbias = (float*)alloc(384 * 4);
    h16* pooled = (h16*)alloc((size_t)784 * 384 * 2);
    h16* keyb   = (h16*)alloc((size_t)784 * 384 * 2);
    h16* valb   = (h16*)alloc((size_t)784 * 384 * 2);
    h16* oqb    = (h16*)alloc((size_t)1024 * 384 * 2);
    h16* gkb    = (h16*)alloc((size_t)1024 * 384 * 2);
    h16* gvb    = (h16*)alloc((size_t)1024 * 384 * 2);
    h16* regA   = (h16*)alloc((size_t)M_TOK * 384 * 2);    // xn -> ow -> x2n
    h16* regB   = (h16*)alloc((size_t)M_TOK * 1536 * 2);   // qkv -> h
    float* x2   = (float*)alloc((size_t)M_TOK * 384 * 4);

    const int T = 256;
    auto cdiv = [](int a, int b) { return (a + b - 1) / b; };

    // --- weight preparation (f32 -> f16 transposed) ---
    transpose_w<<<cdiv(384 * 1152, T), T, 0, stream>>>(qkv_w,   qkvT,   384, 1152);
    transpose_w<<<cdiv(384 * 384,  T), T, 0, stream>>>(proj_w,  projT,  384, 384);
    transpose_w<<<cdiv(384 * 1536, T), T, 0, stream>>>(fc1_w,   fc1T,   384, 1536);
    transpose_w<<<cdiv(1536 * 384, T), T, 0, stream>>>(fc2_w,   fc2T,   1536, 384);
    transpose_w<<<cdiv(384 * 384,  T), T, 0, stream>>>(kproj_w, kprojT, 384, 384);
    transpose_w<<<cdiv(384 * 384,  T), T, 0, stream>>>(vproj_w, vprojT, 384, 384);
    transpose_w<<<cdiv(384 * 384,  T), T, 0, stream>>>(ck_w,    ckT,    384, 384);
    transpose_w<<<cdiv(384 * 384,  T), T, 0, stream>>>(cv_w,    cvT,    384, 384);
    cvt_h16<<<cdiv(64 * 384, T), T, 0, stream>>>(query, queryT, 64 * 384);
    fold_bn<<<2, 192, 0, stream>>>(kproj_b, kbn_g, kbn_b, kgain, kbias);
    fold_bn<<<2, 192, 0, stream>>>(vproj_b, vbn_g, vbn_b, vgain, vbias);

    // --- LN1 + pooling ---
    ln_h16<<<M_TOK, 32, 0, stream>>>(x, n1_g, n1_b, regA);
    pool8<<<784, 384, 0, stream>>>(regA, pooled);

    // --- cross-attention path: key/val projections (conv1x1 + folded BN) ---
    gemm_wmma<EPI_AFFINE_F16><<<dim3(3, 7), 256, 0, stream>>>(
        pooled, kprojT, kbias, kgain, nullptr, keyb, 784, 384, 384);
    gemm_wmma<EPI_AFFINE_F16><<<dim3(3, 7), 256, 0, stream>>>(
        pooled, vprojT, vbias, vgain, nullptr, valb, 784, 384, 384);
    cross_attn<<<dim3(12, 16), 32, 0, stream>>>(queryT, keyb, valb, oqb);
    gemm_wmma<EPI_BIAS_F16><<<dim3(3, 8), 256, 0, stream>>>(
        oqb, ckT, ck_b, nullptr, nullptr, gkb, 1024, 384, 384);
    gemm_wmma<EPI_BIAS_F16><<<dim3(3, 8), 256, 0, stream>>>(
        oqb, cvT, cv_b, nullptr, nullptr, gvb, 1024, 384, 384);

    // --- QKV projection (big GEMM) ---
    gemm_wmma<EPI_BIAS_F16><<<dim3(9, 392), 256, 0, stream>>>(
        regA, qkvT, qkv_b, nullptr, nullptr, regB, M_TOK, 1152, 384);

    // --- windowed attention (+ global kv), writes ow over regA ---
    win_attn<<<dim3(12, 49, 16), 32, 0, stream>>>(regB, gkb, gvb, rpb, regA);

    // --- output projection + shortcut residual -> x2 (f32) ---
    gemm_wmma<EPI_BIAS_RES_F32><<<dim3(3, 392), 256, 0, stream>>>(
        regA, projT, proj_b, nullptr, x, x2, M_TOK, 384, 384);

    // --- LN2 -> regA ---
    ln_h16<<<M_TOK, 32, 0, stream>>>(x2, n2_g, n2_b, regA);

    // --- MLP: fc1 + exact GELU -> regB ---
    gemm_wmma<EPI_BIAS_GELU_F16><<<dim3(12, 392), 256, 0, stream>>>(
        regA, fc1T, fc1_b, nullptr, nullptr, regB, M_TOK, 1536, 384);

    // --- fc2 + residual -> d_out (f32) ---
    gemm_wmma<EPI_BIAS_RES_F32><<<dim3(3, 392), 256, 0, stream>>>(
        regB, fc2T, fc2_b, nullptr, x2, (float*)d_out, M_TOK, 384, 1536);

    (void)in_sizes; (void)n_in; (void)out_size; (void)ws_size;
}
